// Model_41180146434314
// MI455X (gfx1250) — compile-verified, matrix operand order checked by reading, not run on hardware
//
#include <hip/hip_runtime.h>
#include <hip/hip_bf16.h>
#include <math.h>

typedef __attribute__((ext_vector_type(16))) _Float16 v16h;
typedef __attribute__((ext_vector_type(8)))  _Float16 v8h;
typedef __attribute__((ext_vector_type(8)))  float    v8f;
typedef __attribute__((ext_vector_type(4)))  unsigned int u32x4;
typedef __attribute__((ext_vector_type(8)))  unsigned int u32x8;

#define BATCH 512
#define TLEN  256
#define SDIM  512
#define ADIM  128
#define DDIM  64

__device__ __forceinline__ float mishf(float x) {
    float sp = (x > 20.0f) ? x : log1pf(__expf(x));
    return x * tanhf(sp);
}

// ---------------------------------------------------------------------------
// TDM helper: async-load a BM x BK f16 tile (row stride K elements) from
// global into LDS with per-row padding so the LDS stride is AST halves.
//   group0: count=1 | lds_addr | global_addr[56:0] | type=2
//   group1: data_size=2B, pad_enable, pad_interval=16 DW (64B = one row),
//           pad_amount=4 DW (16B -> stride 80B), tile 32 x 128,
//           tensor_dim0_stride = K
// Issued by wave 0 only (TDM ignores EXEC; one issue per wave). Tracked by
// TENSORcnt, consumed via s_wait_tensorcnt.
// ---------------------------------------------------------------------------
__device__ __forceinline__ u32x8 tdm_make_g1(int M, int K) {
    u32x8 g1;
    g1[0] = (1u << 16) | (1u << 20) | (3u << 22) | (3u << 25);
    g1[1] = ((unsigned)K & 0xFFFFu) << 16;                       // dim0 lo
    g1[2] = ((unsigned)K >> 16) | (((unsigned)M & 0xFFFFu) << 16); // dim0 hi | dim1 lo
    g1[3] = ((unsigned)M >> 16) | (32u << 16);                   // dim1 hi | tile_dim0
    g1[4] = 128u;                                                // tile_dim1 | tile_dim2=0
    g1[5] = (unsigned)K;                                         // dim0_stride lo
    g1[6] = 0u;                                                  // dim0_stride hi | dim1_stride lo
    g1[7] = 0u;
    return g1;
}

__device__ __forceinline__ void tdm_load_tile(unsigned lds_addr,
                                              const _Float16* gptr,
                                              u32x8 g1) {
    unsigned long long ga = (unsigned long long)(const void*)gptr;
    u32x4 g0;
    g0[0] = 1u;                                                  // count=1
    g0[1] = lds_addr;
    g0[2] = (unsigned)ga;
    g0[3] = (unsigned)((ga >> 32) & 0x1FFFFFFull) | (2u << 30);  // type=2 (image)
    asm volatile("tensor_load_to_lds %0, %1" :: "s"(g0), "s"(g1) : "memory");
}

// ---------------------------------------------------------------------------
// WMMA GEMM: C[M,N] = act(A[M,K] @ B[K,N] + bias)
// Block tile 128x64, K step 32, 256 threads = 8 waves; wave w owns rows
// [16w,16w+16) x all 64 cols (4 accumulators).
// A_F32 path: stage+convert through VGPRs (projections, x is f32).
// f16 path:   TDM double-buffered A tiles, one barrier per K step.
// ---------------------------------------------------------------------------
#define BM 128
#define BN 64
#define BK 32
#define AST 40   // LDS row stride (halves): 80B, multiple of 16B
#define BST 40

template<int A_F32, int C_F32, int ACT>
__global__ __launch_bounds__(256) void gemm_wmma_kernel(
    const void* __restrict__ Aptr, const float* __restrict__ Bw,
    const float* __restrict__ bias, void* __restrict__ Cptr,
    int M, int N, int K)
{
    __shared__ __align__(16) _Float16 As[2][BM * AST];
    __shared__ __align__(16) _Float16 BsT[2][BN * BST];

    const int tid  = threadIdx.x;
    const int wave = tid >> 5;
    const int lane = tid & 31;
    const int bm = blockIdx.x * BM;
    const int bn = blockIdx.y * BN;

    v8f acc[4] = {};
    const int arow = wave * 16 + (lane & 15);
    const int kb   = (lane >> 4) * 8;

    if constexpr (!A_F32) {
        // ---------------- TDM double-buffered pipeline ----------------
        const _Float16* A = (const _Float16*)Aptr;
        const u32x8 g1 = tdm_make_g1(M, K);
        const unsigned lds0 = (unsigned)(unsigned long long)(void*)&As[0][0];
        const unsigned lds1 = (unsigned)(unsigned long long)(void*)&As[1][0];

        if (tid < 32) tdm_load_tile(lds0, A + (size_t)bm * K, g1);

        int buf = 0;
        for (int k0 = 0; k0 < K; k0 += BK, buf ^= 1) {
            // stage B tile (32 x 64 f32 -> transposed f16), all threads
            {
                const int kk = tid >> 3;
                const int n0 = (tid & 7) * 8;
                const float* src = Bw + (size_t)(k0 + kk) * N + bn + n0;
                if (k0 + BK < K) __builtin_prefetch(src + (size_t)BK * N, 0, 1);
                float f[8];
                *(float4*)(f + 0) = *(const float4*)(src + 0);
                *(float4*)(f + 4) = *(const float4*)(src + 4);
#pragma unroll
                for (int i = 0; i < 8; ++i)
                    BsT[buf][(n0 + i) * BST + kk] = (_Float16)f[i];
            }
            if (tid < 32) __builtin_amdgcn_s_wait_tensorcnt(0);
            __syncthreads();
            // prefetch next A tile into the other buffer (safe: all waves
            // finished reading it before the barrier of this iteration)
            if (tid < 32 && (k0 + BK) < K)
                tdm_load_tile(buf ? lds0 : lds1,
                              A + (size_t)bm * K + (k0 + BK), g1);

            union { v16h v; v8h h[2]; } af;
            af.h[0] = *(const v8h*)&As[buf][arow * AST + kb];
            af.h[1] = *(const v8h*)&As[buf][arow * AST + kb + 16];
#pragma unroll
            for (int nt = 0; nt < 4; ++nt) {
                const int bcol = nt * 16 + (lane & 15);
                union { v16h v; v8h h[2]; } bf;
                bf.h[0] = *(const v8h*)&BsT[buf][bcol * BST + kb];
                bf.h[1] = *(const v8h*)&BsT[buf][bcol * BST + kb + 16];
                acc[nt] = __builtin_amdgcn_wmma_f32_16x16x32_f16(
                    false, af.v, false, bf.v, (short)0, acc[nt], false, false);
            }
        }
    } else {
        // ---------------- f32-A staging path (projections) ----------------
        for (int k0 = 0; k0 < K; k0 += BK) {
            {
                const int r  = tid >> 1;
                const int c0 = (tid & 1) * 16;
                const float* Af = (const float*)Aptr;
                const float* src = Af + (size_t)(bm + r) * K + k0 + c0;
                if (k0 + BK < K) __builtin_prefetch(src + BK, 0, 1);
                float f[16];
                *(float4*)(f + 0)  = *(const float4*)(src + 0);
                *(float4*)(f + 4)  = *(const float4*)(src + 4);
                *(float4*)(f + 8)  = *(const float4*)(src + 8);
                *(float4*)(f + 12) = *(const float4*)(src + 12);
                v8h h0, h1;
#pragma unroll
                for (int i = 0; i < 8; ++i) { h0[i] = (_Float16)f[i]; h1[i] = (_Float16)f[i + 8]; }
                *(v8h*)&As[0][r * AST + c0]     = h0;
                *(v8h*)&As[0][r * AST + c0 + 8] = h1;
            }
            {
                const int kk = tid >> 3;
                const int n0 = (tid & 7) * 8;
                const float* src = Bw + (size_t)(k0 + kk) * N + bn + n0;
                float f[8];
                *(float4*)(f + 0) = *(const float4*)(src + 0);
                *(float4*)(f + 4) = *(const float4*)(src + 4);
#pragma unroll
                for (int i = 0; i < 8; ++i)
                    BsT[0][(n0 + i) * BST + kk] = (_Float16)f[i];
            }
            __syncthreads();

            union { v16h v; v8h h[2]; } af;
            af.h[0] = *(const v8h*)&As[0][arow * AST + kb];
            af.h[1] = *(const v8h*)&As[0][arow * AST + kb + 16];
#pragma unroll
            for (int nt = 0; nt < 4; ++nt) {
                const int bcol = nt * 16 + (lane & 15);
                union { v16h v; v8h h[2]; } bf;
                bf.h[0] = *(const v8h*)&BsT[0][bcol * BST + kb];
                bf.h[1] = *(const v8h*)&BsT[0][bcol * BST + kb + 16];
                acc[nt] = __builtin_amdgcn_wmma_f32_16x16x32_f16(
                    false, af.v, false, bf.v, (short)0, acc[nt], false, false);
            }
            __syncthreads();
        }
    }

    // ---- epilogue: bias + activation + store ----
#pragma unroll
    for (int nt = 0; nt < 4; ++nt) {
        const int n = bn + nt * 16 + (lane & 15);
        const float bv = bias ? bias[n] : 0.0f;
#pragma unroll
        for (int r = 0; r < 8; ++r) {
            const int m = bm + wave * 16 + r + ((lane >> 4) * 8);
            float val = acc[nt][r] + bv;
            if constexpr (ACT) val = mishf(val);
            if constexpr (C_F32) ((float*)Cptr)[(size_t)m * N + n] = val;
            else                 ((_Float16*)Cptr)[(size_t)m * N + n] = (_Float16)val;
        }
    }
}

// ---------------------------------------------------------------------------
// Attention: per-batch block. score = q k^T (256x256) via WMMA, two-pass
// softmax (recompute), colsum over rows, then w = v * colsum, all in LDS.
// ---------------------------------------------------------------------------
__global__ __launch_bounds__(256) void attn_kernel(
    const _Float16* __restrict__ q, const _Float16* __restrict__ k,
    const _Float16* __restrict__ v, _Float16* __restrict__ w)
{
    __shared__ __align__(16) _Float16 qs[TLEN * 72];
    __shared__ __align__(16) _Float16 ks[TLEN * 72];
    __shared__ float rowsum[TLEN];
    __shared__ float colsum[TLEN];

    const int b = blockIdx.x;
    const int tid  = threadIdx.x;
    const int wave = tid >> 5;
    const int lane = tid & 31;

    {
        const _Float16* qg = q + (size_t)b * TLEN * DDIM;
        const _Float16* kg = k + (size_t)b * TLEN * DDIM;
        const int r = tid;
#pragma unroll
        for (int c = 0; c < DDIM; c += 8) {
            *(v8h*)&qs[r * 72 + c] = *(const v8h*)&qg[r * DDIM + c];
            *(v8h*)&ks[r * 72 + c] = *(const v8h*)&kg[r * DDIM + c];
        }
        rowsum[tid] = 0.0f;
        colsum[tid] = 0.0f;
    }
    __syncthreads();

    const int kb = (lane >> 4) * 8;

    // pass 1: rowsum[t] = sum_s exp(score[t,s])
    for (int rt = wave; rt < 16; rt += 8) {
        const int arow = rt * 16 + (lane & 15);
        union { v16h v; v8h h[2]; } a0, a1;
        a0.h[0] = *(const v8h*)&qs[arow * 72 + kb];
        a0.h[1] = *(const v8h*)&qs[arow * 72 + kb + 16];
        a1.h[0] = *(const v8h*)&qs[arow * 72 + 32 + kb];
        a1.h[1] = *(const v8h*)&qs[arow * 72 + 32 + kb + 16];
        for (int nt = 0; nt < 16; ++nt) {
            const int bcol = nt * 16 + (lane & 15);
            union { v16h v; v8h h[2]; } b0, b1;
            b0.h[0] = *(const v8h*)&ks[bcol * 72 + kb];
            b0.h[1] = *(const v8h*)&ks[bcol * 72 + kb + 16];
            b1.h[0] = *(const v8h*)&ks[bcol * 72 + 32 + kb];
            b1.h[1] = *(const v8h*)&ks[bcol * 72 + 32 + kb + 16];
            v8f acc = {};
            acc = __builtin_amdgcn_wmma_f32_16x16x32_f16(false, a0.v, false, b0.v, (short)0, acc, false, false);
            acc = __builtin_amdgcn_wmma_f32_16x16x32_f16(false, a1.v, false, b1.v, (short)0, acc, false, false);
#pragma unroll
            for (int r = 0; r < 8; ++r) {
                const int row = rt * 16 + r + ((lane >> 4) * 8);
                atomicAdd(&rowsum[row], __expf(acc[r]));
            }
        }
    }
    __syncthreads();

    // pass 2: colsum[s] = sum_t exp(score[t,s]) / rowsum[t]
    for (int rt = wave; rt < 16; rt += 8) {
        const int arow = rt * 16 + (lane & 15);
        union { v16h v; v8h h[2]; } a0, a1;
        a0.h[0] = *(const v8h*)&qs[arow * 72 + kb];
        a0.h[1] = *(const v8h*)&qs[arow * 72 + kb + 16];
        a1.h[0] = *(const v8h*)&qs[arow * 72 + 32 + kb];
        a1.h[1] = *(const v8h*)&qs[arow * 72 + 32 + kb + 16];
        for (int nt = 0; nt < 16; ++nt) {
            const int bcol = nt * 16 + (lane & 15);
            union { v16h v; v8h h[2]; } b0, b1;
            b0.h[0] = *(const v8h*)&ks[bcol * 72 + kb];
            b0.h[1] = *(const v8h*)&ks[bcol * 72 + kb + 16];
            b1.h[0] = *(const v8h*)&ks[bcol * 72 + 32 + kb];
            b1.h[1] = *(const v8h*)&ks[bcol * 72 + 32 + kb + 16];
            v8f acc = {};
            acc = __builtin_amdgcn_wmma_f32_16x16x32_f16(false, a0.v, false, b0.v, (short)0, acc, false, false);
            acc = __builtin_amdgcn_wmma_f32_16x16x32_f16(false, a1.v, false, b1.v, (short)0, acc, false, false);
            float csum = 0.0f;
#pragma unroll
            for (int r = 0; r < 8; ++r) {
                const int row = rt * 16 + r + ((lane >> 4) * 8);
                csum += __expf(acc[r]) / rowsum[row];
            }
            atomicAdd(&colsum[nt * 16 + (lane & 15)], csum);
        }
    }
    __syncthreads();

    {
        const _Float16* vg = v + (size_t)b * TLEN * DDIM;
        _Float16* wg = w + (size_t)b * TLEN * DDIM;
        for (int i = tid; i < TLEN * DDIM; i += 256) {
            const int t = i >> 6;
            wg[i] = (_Float16)((float)vg[i] * colsum[t]);
        }
    }
}

// ---------------------------------------------------------------------------
__global__ void head2_kernel(const _Float16* __restrict__ g,
                             const float* __restrict__ w22,
                             const float* __restrict__ b22,
                             float* __restrict__ out)
{
    const int row = blockIdx.x * blockDim.x + threadIdx.x;
    if (row >= BATCH) return;
    float s = 0.0f;
    for (int i = 0; i < 256; ++i)
        s += (float)g[row * 256 + i] * w22[i];
    s += b22[0];
    out[row] = 1.0f / (1.0f + __expf(-s));
}

// ---------------------------------------------------------------------------
extern "C" void kernel_launch(void* const* d_in, const int* in_sizes, int n_in,
                              void* d_out, int out_size, void* d_ws, size_t ws_size,
                              hipStream_t stream) {
    const float* x   = (const float*)d_in[0];
    const float* wq  = (const float*)d_in[1];
    const float* bq  = (const float*)d_in[2];
    const float* wk  = (const float*)d_in[3];
    const float* bk  = (const float*)d_in[4];
    const float* wv  = (const float*)d_in[5];
    const float* bv  = (const float*)d_in[6];
    const float* w11 = (const float*)d_in[7];
    const float* b11 = (const float*)d_in[8];
    const float* w12 = (const float*)d_in[9];
    const float* b12 = (const float*)d_in[10];
    const float* w13 = (const float*)d_in[11];
    const float* b13 = (const float*)d_in[12];
    const float* w21 = (const float*)d_in[13];
    const float* b21 = (const float*)d_in[14];
    const float* w22 = (const float*)d_in[15];
    const float* b22 = (const float*)d_in[16];

    char* ws = (char*)d_ws;
    const size_t M = (size_t)BATCH * TLEN;  // 131072
    _Float16* qf = (_Float16*)(ws);
    _Float16* kf = (_Float16*)(ws + M * DDIM * 2);
    _Float16* vf = (_Float16*)(ws + 2 * M * DDIM * 2);
    _Float16* wf = (_Float16*)(ws + 3 * M * DDIM * 2);
    _Float16* h1 = (_Float16*)(ws + 4 * M * DDIM * 2);
    _Float16* gg = (_Float16*)(ws + 4 * M * DDIM * 2 + 512u * 256 * 2);
    _Float16* h2 = (_Float16*)(ws + 4 * M * DDIM * 2 + 2u * 512 * 256 * 2);

    float* q_val = (float*)d_out;
    float* n_out = (float*)d_out + BATCH * ADIM;

    // 1) projections (A = x, f32): M=131072, K=512, N=64
    gemm_wmma_kernel<1, 0, 0><<<dim3(M / BM, DDIM / BN), 256, 0, stream>>>(
        x, wq, bq, qf, (int)M, DDIM, SDIM);
    gemm_wmma_kernel<1, 0, 0><<<dim3(M / BM, DDIM / BN), 256, 0, stream>>>(
        x, wk, bk, kf, (int)M, DDIM, SDIM);
    gemm_wmma_kernel<1, 0, 0><<<dim3(M / BM, DDIM / BN), 256, 0, stream>>>(
        x, wv, bv, vf, (int)M, DDIM, SDIM);

    // 2) attention colsum + w = v * colsum
    attn_kernel<<<BATCH, 256, 0, stream>>>(qf, kf, vf, wf);

    // 3) h1 = mish(w@w11+b11), g = mish(w@w21+b21) (TDM-fed A tiles)
    gemm_wmma_kernel<0, 0, 1><<<dim3(BATCH / BM, 256 / BN), 256, 0, stream>>>(
        wf, w11, b11, h1, BATCH, 256, TLEN * DDIM);
    gemm_wmma_kernel<0, 0, 1><<<dim3(BATCH / BM, 256 / BN), 256, 0, stream>>>(
        wf, w21, b21, gg, BATCH, 256, TLEN * DDIM);

    // 4) h2 = mish(h1@w12+b12)
    gemm_wmma_kernel<0, 0, 1><<<dim3(BATCH / BM, 512 / BN), 256, 0, stream>>>(
        h1, w12, b12, h2, BATCH, 512, 256);

    // 5) q_val = mish(h2@w13+b13) -> d_out (fp32)
    gemm_wmma_kernel<0, 1, 1><<<dim3(BATCH / BM, ADIM / BN), 256, 0, stream>>>(
        h2, w13, b13, q_val, BATCH, ADIM, SDIM);

    // 6) n = sigmoid(g@w22+b22)
    head2_kernel<<<dim3(BATCH / 256), 256, 0, stream>>>(gg, w22, b22, n_out);
}
